// Rig2DConv_85925115723874
// MI455X (gfx1250) — matrix operand order repaired and checked by reading, not compiled, not run on hardware
//
#include <hip/hip_runtime.h>
#include <hip/hip_bf16.h>

// MI455X (gfx1250) steerable-filter conv + per-angle bilinear rotation.
//
// Pass 1: prep   -- build 32 Gaussian-derivative kernels, pre-packed into the
//                   wave32 WMMA B-fragment layout (f16, K padded 81->96).
// Pass 2: conv   -- im2col GEMM per 16x8 output tile using
//                   v_wmma_f32_16x16x32_f16 (M=16 pixels, N=16 ch x2, K=32 x3).
//                   All tap->LDS offsets are compile-time; only two runtime
//                   base registers per wave (group*8 / group*23 deltas).
// Pass 3: rotate -- bilinear rotation gather; y (128MB) stays resident in the
//                   192MB L2, so scattered reads do not hit HBM again.

typedef __attribute__((ext_vector_type(16))) _Float16 v16h;
typedef __attribute__((ext_vector_type(8)))  float    v8f;

#define KS     9
#define PHI_N  8
#define F_N    4
#define OC     32         // F*PHI output channels
#define H_IMG  256
#define W_IMG  256
#define B_IMG  16
#define PI_F   3.14159265358979f

// ws layout: [0, 6144) packed B fragments (6 frags * 32 lanes * 16 halves),
//            y buffer at byte offset 8192 (16*256*256*32 f32 = 128 MiB).
#define Y_OFF_BYTES 8192

// ---------------------------------------------------------------------------
// Pass 1: build kernels, packed per-lane for the WMMA B operand.
// Fragment fi = nt*3 + kb  (nt: N-tile 0..1, kb: K-block 0..2).
// Element e of lane's v16h corresponds to K = kb*32 + (e/8)*16 + (lane/16)*8 + (e%8),
// channel oc = nt*16 + (lane%16).  K in [81,96) is zero padding.
// ---------------------------------------------------------------------------
__global__ __launch_bounds__(256)
void prep_kernel(const float* __restrict__ l, const float* __restrict__ alpha,
                 const float* __restrict__ beta, _Float16* __restrict__ kp) {
  for (int idx = threadIdx.x; idx < 6 * 32 * 16; idx += 256) {
    int fi   = idx >> 9;          // / (32*16)
    int lane = (idx >> 4) & 31;
    int e    = idx & 15;
    int nt = fi / 3, kb = fi % 3;
    int group = lane >> 4, n = lane & 15;
    int K  = kb * 32 + (e >> 3) * 16 + group * 8 + (e & 7);
    int oc = nt * 16 + n;
    int f = oc >> 3, phi = oc & 7;
    float val = 0.0f;
    if (K < KS * KS) {
      int ky = K / KS, kx = K % KS;
      float L = l[f], a = alpha[f], b = beta[f];
      float xg = (float)kx - 4.0f;   // CENTER = 4
      float yg = (float)ky - 4.0f;
      float E   = expf(-L * (a * xg * xg + b * yg * yg));
      float arx = -2.0f * a * L * L * xg * (1.0f / PI_F) * E;
      float ary = -2.0f * b * L * L * yg * (1.0f / PI_F) * E;
      float ang = (float)phi * (PI_F * 0.25f);  // linspace(0,360,9)[:8] deg
      val = cosf(ang) * arx + sinf(ang) * ary;
    }
    kp[idx] = (_Float16)val;
  }
}

// ---------------------------------------------------------------------------
// Pass 2: convolution via WMMA. Block = 256 threads = 8 wave32.
// Block tile: 16 columns x 8 rows; wave w handles output row r0+w.
// ISA 16-bit A layout: element e of lane L -> K = (e/8)*16 + (L/16)*8 + (e%8).
// LDS tap offset off(K) = (K/9)*24 + K%9; off(K0+8)-off(K0) is a compile-time
// constant in {8, 23}, so the gather needs no runtime div/mod at all.
// ---------------------------------------------------------------------------
__global__ __launch_bounds__(256)
void conv_kernel(const float* __restrict__ x, const v16h* __restrict__ kp,
                 float* __restrict__ y) {
  __shared__ _Float16 tile[16 * 24];

  const int tid = threadIdx.x;
  const int c0  = blockIdx.x * 16;
  const int r0  = blockIdx.y * 8;
  const int b   = blockIdx.z;

  // Stage halo tile (rows r0-4..r0+11, cols c0-4..c0+19) as f16, zero-fill OOB.
  for (int i = tid; i < 16 * 24; i += 256) {
    int ty = i / 24, tx = i % 24;
    int gy = r0 - 4 + ty, gx = c0 - 4 + tx;
    float v = 0.0f;
    if (gy >= 0 && gy < H_IMG && gx >= 0 && gx < W_IMG)
      v = x[((size_t)b * H_IMG + gy) * W_IMG + gx];
    tile[i] = (_Float16)v;
  }
  __syncthreads();

  const int wv    = tid >> 5;    // wave id -> output row r0+wv
  const int lane  = tid & 31;
  const int group = lane >> 4;
  const int m     = lane & 15;   // A-matrix M (pixel) == B-matrix N (channel)

  // B fragments: one 32-byte vector load each from the pre-packed buffer.
  v16h bf0[3], bf1[3];
#pragma unroll
  for (int kb = 0; kb < 3; ++kb) {
    bf0[kb] = kp[(0 * 3 + kb) * 32 + lane];
    bf1[kb] = kp[(1 * 3 + kb) * 32 + lane];
  }

  // Two runtime gather bases; every element offset is an immediate.
  const int base0  = wv * 24 + m;
  const int gsel8  = base0 + group * 8;    // for elements with K0%9 == 0
  const int gsel23 = base0 + group * 23;   // for elements with K0%9 != 0

  v8f acc0 = {};
  v8f acc1 = {};
#pragma unroll
  for (int kb = 0; kb < 3; ++kb) {
    v16h a;
#pragma unroll
    for (int e = 0; e < 16; ++e) {
      const int K0 = kb * 32 + (e >> 3) * 16 + (e & 7);  // group-0 tap id
      if (kb == 2 && e > 8) {
        // K >= 81 for both lane groups: compile-time zero padding.
        a[e] = (_Float16)0.0f;
      } else if (kb == 2 && e == 8) {
        // K = 80 for group 0 (valid), 88 for group 1 (padding).
        const int off0 = (K0 / KS) * 24 + (K0 % KS);
        _Float16 v = tile[base0 + off0];
        _Float16 z = (_Float16)0.0f;
        a[e] = group ? z : v;
      } else {
        const int off0  = (K0 / KS) * 24 + (K0 % KS);
        const int off1  = ((K0 + 8) / KS) * 24 + ((K0 + 8) % KS);
        const int delta = off1 - off0;          // compile-time: 8 or 23
        a[e] = tile[(delta == 8 ? gsel8 : gsel23) + off0];
      }
    }
    acc0 = __builtin_amdgcn_wmma_f32_16x16x32_f16(false, a, false, bf0[kb],
                                                  (short)0, acc0, false, false);
    acc1 = __builtin_amdgcn_wmma_f32_16x16x32_f16(false, a, false, bf1[kb],
                                                  (short)0, acc1, false, false);
  }

  // D layout: VGPR r -> M = r + 8*group, N = lane&15.  y is [b][h][w][oc].
  float* yb = y + (((size_t)b * H_IMG + (r0 + wv)) * W_IMG + c0) * OC;
#pragma unroll
  for (int r = 0; r < 8; ++r) {
    int M = r + 8 * group;
    yb[M * OC + m]      = acc0[r];   // channels 0..15
    yb[M * OC + 16 + m] = acc1[r];   // channels 16..31
  }
}

// ---------------------------------------------------------------------------
// Pass 3: per-angle bilinear rotation (CCW about image center, zero fill).
// One thread per (b, phi, yo, xo); writes 4 contiguous f32 (the F dim).
// ---------------------------------------------------------------------------
__global__ __launch_bounds__(256)
void rotate_kernel(const float* __restrict__ y, float* __restrict__ out) {
  unsigned t = blockIdx.x * 256u + threadIdx.x;
  int xo  = t & 255;
  int yo  = (t >> 8) & 255;
  int phi = (t >> 16) & 7;
  int b   = (int)(t >> 19);

  float theta = (float)phi * (PI_F * 0.25f);
  float ct = cosf(theta), st = sinf(theta);
  const float cc = 127.5f;  // (256-1)*0.5

  float xf = (float)xo - cc, yf = (float)yo - cc;
  float xi = ct * xf - st * yf + cc;
  float yi = st * xf + ct * yf + cc;
  float x0 = floorf(xi), y0 = floorf(yi);
  float wx = xi - x0, wy = yi - y0;

  float acc[F_N] = {0.f, 0.f, 0.f, 0.f};
  const float* ybase = y + (size_t)b * H_IMG * W_IMG * OC;

  auto tap = [&](float ys, float xs, float w) {
    bool valid = (ys >= 0.f) && (ys <= 255.f) && (xs >= 0.f) && (xs <= 255.f);
    float wv = valid ? w : 0.f;
    int yc = (int)fminf(fmaxf(ys, 0.f), 255.f);
    int xc = (int)fminf(fmaxf(xs, 0.f), 255.f);
    const float* p = ybase + ((size_t)yc * W_IMG + xc) * OC + phi;
#pragma unroll
    for (int f = 0; f < F_N; ++f) acc[f] += wv * p[f * PHI_N];
  };

  tap(y0,       x0,       (1.f - wy) * (1.f - wx));
  tap(y0,       x0 + 1.f, (1.f - wy) * wx);
  tap(y0 + 1.f, x0,       wy * (1.f - wx));
  tap(y0 + 1.f, x0 + 1.f, wy * wx);

  float* o = out + (size_t)t * F_N;   // (B, PHI, H, W, F) linear
#pragma unroll
  for (int f = 0; f < F_N; ++f) o[f] = acc[f];
}

// ---------------------------------------------------------------------------
extern "C" void kernel_launch(void* const* d_in, const int* in_sizes, int n_in,
                              void* d_out, int out_size, void* d_ws, size_t ws_size,
                              hipStream_t stream) {
  const float* x     = (const float*)d_in[0];
  const float* l     = (const float*)d_in[1];
  const float* alpha = (const float*)d_in[2];
  const float* beta  = (const float*)d_in[3];
  float* out = (float*)d_out;

  _Float16* kp = (_Float16*)d_ws;
  float*    y  = (float*)((char*)d_ws + Y_OFF_BYTES);

  prep_kernel<<<1, 256, 0, stream>>>(l, alpha, beta, kp);

  dim3 cgrid(W_IMG / 16, H_IMG / 8, B_IMG);
  conv_kernel<<<cgrid, 256, 0, stream>>>(x, (const v16h*)kp, y);

  unsigned nthreads = (unsigned)B_IMG * PHI_N * H_IMG * W_IMG;  // 8.4M
  rotate_kernel<<<nthreads / 256, 256, 0, stream>>>(y, out);
}
